// PatchTSTEncoder_85478439125319
// MI455X (gfx1250) — matrix operand order repaired and checked by reading, not compiled, not run on hardware
//
#include <hip/hip_runtime.h>
#include <hip/hip_bf16.h>
#include <math.h>

typedef _Float16 f16;
typedef _Float16 v16h __attribute__((ext_vector_type(16)));
typedef _Float16 v8h  __attribute__((ext_vector_type(8)));
typedef float    v8f  __attribute__((ext_vector_type(8)));
typedef int      v4i  __attribute__((ext_vector_type(4)));

constexpr int E_ = 3, D_ = 512, H_ = 8, F_ = 2048, B_ = 32, V_ = 21, L_ = 128;
constexpr int NSEQ  = B_ * V_;      // 672
constexpr int MROWS = NSEQ * L_;    // 86016 rows of [*, D]
constexpr int DH_   = D_ / H_;      // 64
constexpr float EPS_ = 1e-5f;

// ---------------------------------------------------------------------------
// Async global->LDS copy (CDNA5 GLOBAL_LOAD_ASYNC_TO_LDS_B128, ASYNCcnt),
// with a synchronous float4 fallback when the builtin is unavailable.
// Toolchain signature (from probe): (global int4*, lds int4*, imm, imm)
// ---------------------------------------------------------------------------
#if defined(__has_builtin)
#if __has_builtin(__builtin_amdgcn_global_load_async_to_lds_b128) && \
    __has_builtin(__builtin_amdgcn_s_wait_asynccnt)
#define HAVE_ASYNC_LDS 1
#endif
#endif
#ifndef HAVE_ASYNC_LDS
#define HAVE_ASYNC_LDS 0
#endif

__device__ __forceinline__ void cp16_g2l(f16* dst_lds, const f16* src_glob) {
#if HAVE_ASYNC_LDS
  __builtin_amdgcn_global_load_async_to_lds_b128(
      (__attribute__((address_space(1))) v4i*)src_glob,
      (__attribute__((address_space(3))) v4i*)dst_lds, 0, 0);
#else
  *(float4*)dst_lds = *(const float4*)src_glob;
#endif
}
__device__ __forceinline__ void cp_wait() {
#if HAVE_ASYNC_LDS
  __builtin_amdgcn_s_wait_asynccnt(0);
#endif
}

// A fragment (contiguous K order): v16h elems 0..15 <- K = half*16 + 0..15.
// Two contiguous 16B LDS loads.
__device__ __forceinline__ v16h ld_afrag(const f16* p) {
  v8h lo = *(const v8h*)p;
  v8h hi = *(const v8h*)(p + 8);
  return __builtin_shufflevector(lo, hi, 0, 1, 2, 3, 4, 5, 6, 7,
                                 8, 9, 10, 11, 12, 13, 14, 15);
}

// ---------------------------------------------------------------------------
// f32 -> f16 conversion (activations)
// ---------------------------------------------------------------------------
__global__ void k_cvt(const float* __restrict__ s, f16* __restrict__ d, size_t n) {
  size_t i = (size_t)blockIdx.x * blockDim.x + threadIdx.x;
  if (i < n) d[i] = (f16)s[i];
}

// f32 [K,N] -> f16 [N,K] transposed conversion (weights, once per launch).
// blockIdx.z selects the layer slice; 32x32 LDS tile for coalescing.
__global__ __launch_bounds__(256)
void k_cvt_t(const float* __restrict__ s, f16* __restrict__ d, int K, int N) {
  __shared__ float tile[32][33];
  s += (size_t)blockIdx.z * K * N;
  d += (size_t)blockIdx.z * K * N;
  const int kb = blockIdx.y * 32, nb = blockIdx.x * 32;
  const int tx = threadIdx.x & 31, ty = threadIdx.x >> 5;  // ty in 0..7
  for (int r = ty; r < 32; r += 8)
    tile[r][tx] = s[(size_t)(kb + r) * N + nb + tx];
  __syncthreads();
  for (int r = ty; r < 32; r += 8)
    d[(size_t)(nb + r) * K + kb + tx] = (f16)tile[tx][r];
}

// ---------------------------------------------------------------------------
// WMMA GEMM: C[M,N] = A[M,K] (f16 row-major) x W^T[N,K] (f16, pre-transposed)
// + bias. 128x128 tile, 256 threads = 8 wave32 waves, K-step 32.
// mode 0: store f16   mode 1: store f32   mode 2: exact GELU then store f16
// ---------------------------------------------------------------------------
__global__ __launch_bounds__(256)
void k_gemm(const f16* __restrict__ A, const f16* __restrict__ Wt,
            const float* __restrict__ bias, float* __restrict__ out32,
            f16* __restrict__ out16, int K, int N, int mode) {
  __shared__ alignas(32) f16 sA[128 * 32];   // [row][k]
  __shared__ alignas(32) f16 sBt[128 * 32];  // [col][k]
  const int tid  = threadIdx.x;
  const int wave = tid >> 5, lane = tid & 31;
  const int halfsel = lane >> 4, lr = lane & 15;
  const int m0 = blockIdx.y * 128, n0 = blockIdx.x * 128;
  const int arow = wave * 16 + lr;

  v8f acc[8] = {};

  for (int kt = 0; kt < K; kt += 32) {
    // Stage A tile (128x32) and B tile (128 cols x 32 k) as 16B async chunks
    for (int c = tid; c < 512; c += 256) {
      int idx = c * 8, r = idx >> 5, kk = idx & 31;
      cp16_g2l(&sA[idx],  &A [(size_t)(m0 + r) * K + kt + kk]);
      cp16_g2l(&sBt[idx], &Wt[(size_t)(n0 + r) * K + kt + kk]);
    }
    cp_wait();
    __syncthreads();

    v16h a = ld_afrag(&sA[arow * 32 + halfsel * 16]);
#pragma unroll
    for (int j = 0; j < 8; ++j) {
      // B fragment: lane -> column, elem e -> K = halfsel*16 + e (contiguous)
      v16h b = *(const v16h*)&sBt[(j * 16 + lr) * 32 + halfsel * 16];
      acc[j] = __builtin_amdgcn_wmma_f32_16x16x32_f16(
          false, a, false, b, (short)0, acc[j], false, false);
    }
    __syncthreads();
  }

  // Epilogue per C layout: VGPR r -> row 8*(lane/16)+r, lane%16 -> column
#pragma unroll
  for (int j = 0; j < 8; ++j) {
    int col = n0 + j * 16 + lr;
    float bv = bias[col];
#pragma unroll
    for (int r = 0; r < 8; ++r) {
      int row = m0 + wave * 16 + halfsel * 8 + r;
      float v = acc[j][r] + bv;
      if (mode == 2) v = 0.5f * v * (1.0f + erff(v * 0.70710678118654752f));
      if (mode == 1) out32[(size_t)row * N + col] = v;
      else           out16[(size_t)row * N + col] = (f16)v;
    }
  }
}

// ---------------------------------------------------------------------------
// Fused attention per (sequence, head): scores = QK^T/8 (+prev, written back),
// softmax, ctx = P V. One block of 8 waves; all tiles resident in LDS.
// ---------------------------------------------------------------------------
__global__ __launch_bounds__(256)
void k_attn(const f16* __restrict__ Qg, const f16* __restrict__ Kg,
            const f16* __restrict__ Vg, float* __restrict__ scoresG,
            f16* __restrict__ ctxG, int add_prev) {
  extern __shared__ char smem[];
  f16*   sQ  = (f16*)smem;               // 128 x 64  [l][d]
  f16*   sK  = sQ + 128 * 64;            // 128 x 64  [m][d] (== K^T staging)
  f16*   sVt = sK + 128 * 64;            // 64 x 128  [d][l] (V transposed)
  float* sS  = (float*)(sVt + 64 * 128); // 128 x 128 f32 scores
  f16*   sP  = (f16*)(sS + 128 * 128);   // 128 x 128 f16 probabilities

  const int tid  = threadIdx.x;
  const int wave = tid >> 5, lane = tid & 31;
  const int halfsel = lane >> 4, lr = lane & 15;
  const int n = blockIdx.x / H_, h = blockIdx.x % H_;
  const size_t base = (size_t)n * L_ * D_ + (size_t)h * DH_;
  const int arow = wave * 16 + lr;

  // Stage Q/K (async, contiguous) and V (transposed scatter into LDS)
  for (int c = tid; c < 1024; c += 256) {
    int idx = c * 8, l = idx >> 6, dd = idx & 63;
    size_t g = base + (size_t)l * D_ + dd;
    cp16_g2l(&sQ[idx], &Qg[g]);
    cp16_g2l(&sK[idx], &Kg[g]);
    float4 t = *(const float4*)&Vg[g];
    f16 tmp[8];
    *(float4*)tmp = t;
#pragma unroll
    for (int e2 = 0; e2 < 8; ++e2) sVt[(dd + e2) * 128 + l] = tmp[e2];
  }
  cp_wait();
  __syncthreads();

  const size_t sbase = (size_t)blockIdx.x * L_ * L_;

  { // scores = Q x K^T : K-dim = DH=64 -> two K=32 WMMA steps per tile
    v8f acc[8] = {};
    for (int kt = 0; kt < 64; kt += 32) {
      v16h a = ld_afrag(&sQ[arow * 64 + kt + halfsel * 16]);
#pragma unroll
      for (int j = 0; j < 8; ++j) {
        // B[d,m] = K[m,d]: contiguous in d within sK row m
        v16h b = *(const v16h*)&sK[(j * 16 + lr) * 64 + kt + halfsel * 16];
        acc[j] = __builtin_amdgcn_wmma_f32_16x16x32_f16(
            false, a, false, b, (short)0, acc[j], false, false);
      }
    }
#pragma unroll
    for (int j = 0; j < 8; ++j)
#pragma unroll
      for (int r = 0; r < 8; ++r) {
        int row = wave * 16 + halfsel * 8 + r, col = j * 16 + lr;
        float s = acc[j][r] * 0.125f;  // 1/sqrt(DH)
        if (add_prev) s += scoresG[sbase + (size_t)row * L_ + col];
        scoresG[sbase + (size_t)row * L_ + col] = s;  // RealFormer carry
        sS[row * 128 + col] = s;
      }
  }
  __syncthreads();

  // Row softmax, emit f16 probabilities
  if (tid < 128) {
    float mx = -1e30f;
    for (int c = 0; c < 128; ++c) mx = fmaxf(mx, sS[tid * 128 + c]);
    float sum = 0.f;
    for (int c = 0; c < 128; ++c) {
      float p = __expf(sS[tid * 128 + c] - mx);
      sS[tid * 128 + c] = p;
      sum += p;
    }
    float inv = 1.0f / sum;
    for (int c = 0; c < 128; ++c) sP[tid * 128 + c] = (f16)(sS[tid * 128 + c] * inv);
  }
  __syncthreads();

  { // ctx = P x V : 128x128 @ 128x64 -> four K=32 steps, 4 column tiles
    v8f acc[4] = {};
    for (int kt = 0; kt < 128; kt += 32) {
      v16h a = ld_afrag(&sP[arow * 128 + kt + halfsel * 16]);
#pragma unroll
      for (int j = 0; j < 4; ++j) {
        // B[k,d] = V[k][d] = sVt[d][k]: contiguous in k
        v16h b = *(const v16h*)&sVt[(j * 16 + lr) * 128 + kt + halfsel * 16];
        acc[j] = __builtin_amdgcn_wmma_f32_16x16x32_f16(
            false, a, false, b, (short)0, acc[j], false, false);
      }
    }
#pragma unroll
    for (int j = 0; j < 4; ++j)
#pragma unroll
      for (int r = 0; r < 8; ++r) {
        int row = wave * 16 + halfsel * 8 + r, dcol = j * 16 + lr;
        ctxG[base + (size_t)row * D_ + dcol] = (f16)acc[j][r];
      }
  }
}

// ---------------------------------------------------------------------------
// BatchNorm stats over (N,L) per channel of (x + y); biased variance
// ---------------------------------------------------------------------------
__global__ __launch_bounds__(256)
void k_bnstats(const float* __restrict__ x, const float* __restrict__ y,
               float* __restrict__ mean, float* __restrict__ var) {
  __shared__ float s1[256], s2[256];
  const int c = blockIdx.x;
  float sa = 0.f, sb = 0.f;
  for (int i = threadIdx.x; i < MROWS; i += 256) {
    float v = x[(size_t)i * D_ + c] + y[(size_t)i * D_ + c];
    sa += v; sb += v * v;
  }
  s1[threadIdx.x] = sa; s2[threadIdx.x] = sb;
  __syncthreads();
  for (int st = 128; st > 0; st >>= 1) {
    if (threadIdx.x < st) {
      s1[threadIdx.x] += s1[threadIdx.x + st];
      s2[threadIdx.x] += s2[threadIdx.x + st];
    }
    __syncthreads();
  }
  if (threadIdx.x == 0) {
    float m = s1[0] / (float)MROWS;
    mean[c] = m;
    var[c]  = s2[0] / (float)MROWS - m * m;
  }
}

// Residual add + normalize + affine; refresh f32 master and f16 mirror
__global__ void k_bnapply(const float* __restrict__ x, const float* __restrict__ y,
                          const float* __restrict__ mean, const float* __restrict__ var,
                          const float* __restrict__ g, const float* __restrict__ be,
                          float* __restrict__ out32, f16* __restrict__ out16) {
  size_t i = (size_t)blockIdx.x * blockDim.x + threadIdx.x;
  if (i >= (size_t)MROWS * D_) return;
  int c = (int)(i % D_);
  float v = (x[i] + y[i] - mean[c]) * rsqrtf(var[c] + EPS_) * g[c] + be[c];
  out32[i] = v;
  out16[i] = (f16)v;
}

// ---------------------------------------------------------------------------
extern "C" void kernel_launch(void* const* d_in, const int* in_sizes, int n_in,
                              void* d_out, int out_size, void* d_ws, size_t ws_size,
                              hipStream_t stream) {
  (void)in_sizes; (void)n_in; (void)out_size; (void)ws_size;
  const float* x  = (const float*)d_in[0];
  const float* Wq = (const float*)d_in[1];  const float* bq = (const float*)d_in[2];
  const float* Wk = (const float*)d_in[3];  const float* bk = (const float*)d_in[4];
  const float* Wv = (const float*)d_in[5];  const float* bv = (const float*)d_in[6];
  const float* Wo = (const float*)d_in[7];  const float* bo = (const float*)d_in[8];
  const float* W1 = (const float*)d_in[9];  const float* b1 = (const float*)d_in[10];
  const float* W2 = (const float*)d_in[11]; const float* b2 = (const float*)d_in[12];
  const float* g1 = (const float*)d_in[13]; const float* be1 = (const float*)d_in[14];
  const float* g2 = (const float*)d_in[15]; const float* be2 = (const float*)d_in[16];

  char* ws = (char*)d_ws;
  size_t off = 0;
  auto carve = [&](size_t bytes) -> char* {
    char* p = ws + off;
    off += (bytes + 255) & ~(size_t)255;
    return p;
  };
  const size_t MD = (size_t)MROWS * D_;
  const size_t MF = (size_t)MROWS * F_;
  float* hf32   = (float*)carve(MD * 4);
  float* t32    = (float*)carve(MD * 4);
  f16*   hf16   = (f16*)carve(MD * 2);
  f16*   qf16   = (f16*)carve(MD * 2);
  f16*   kf16   = (f16*)carve(MD * 2);
  f16*   vf16   = (f16*)carve(MD * 2);
  f16*   ctx16  = (f16*)carve(MD * 2);
  f16*   ffh16  = (f16*)carve(MF * 2);
  float* scores = (float*)carve((size_t)NSEQ * H_ * L_ * L_ * 4);
  f16*   wq16   = (f16*)carve((size_t)E_ * D_ * D_ * 2);  // all stored [N,K]
  f16*   wk16   = (f16*)carve((size_t)E_ * D_ * D_ * 2);
  f16*   wv16   = (f16*)carve((size_t)E_ * D_ * D_ * 2);
  f16*   wo16   = (f16*)carve((size_t)E_ * D_ * D_ * 2);
  f16*   w116   = (f16*)carve((size_t)E_ * D_ * F_ * 2);
  f16*   w216   = (f16*)carve((size_t)E_ * F_ * D_ * 2);
  float* mean   = (float*)carve(D_ * 4);
  float* var    = (float*)carve(D_ * 4);

  // Activations: fp32 master + fp16 mirror
  (void)hipMemcpyAsync(hf32, x, MD * 4, hipMemcpyDeviceToDevice, stream);
  k_cvt<<<dim3((unsigned)((MD + 255) / 256)), 256, 0, stream>>>(x, hf16, MD);

  // Weights: convert + transpose once ([K,N] f32 -> [N,K] f16); ~19 MB total,
  // stays resident in the 192 MB L2 for the whole pass.
  auto cvtT = [&](const float* s, f16* d, int K, int N) {
    dim3 g(N / 32, K / 32, E_);
    k_cvt_t<<<g, 256, 0, stream>>>(s, d, K, N);
  };
  cvtT(Wq, wq16, D_, D_);
  cvtT(Wk, wk16, D_, D_);
  cvtT(Wv, wv16, D_, D_);
  cvtT(Wo, wo16, D_, D_);
  cvtT(W1, w116, D_, F_);
  cvtT(W2, w216, F_, D_);

  auto gemm = [&](const f16* A, const f16* Wt, const float* bias,
                  int K, int N, int mode, float* o32, f16* o16) {
    dim3 grid(N / 128, MROWS / 128);
    k_gemm<<<grid, 256, 0, stream>>>(A, Wt, bias, o32, o16, K, N, mode);
  };

  const unsigned smem_attn =
      2u * 128 * 64 * 2 + 64u * 128 * 2 + 128u * 128 * 4 + 128u * 128 * 2;

  for (int e = 0; e < E_; ++e) {
    const f16* wq = wq16 + (size_t)e * D_ * D_;
    const f16* wk = wk16 + (size_t)e * D_ * D_;
    const f16* wv = wv16 + (size_t)e * D_ * D_;
    const f16* wo = wo16 + (size_t)e * D_ * D_;
    const f16* w1 = w116 + (size_t)e * D_ * F_;
    const f16* w2 = w216 + (size_t)e * F_ * D_;

    gemm(hf16, wq, bq + e * D_, D_, D_, 0, nullptr, qf16);
    gemm(hf16, wk, bk + e * D_, D_, D_, 0, nullptr, kf16);
    gemm(hf16, wv, bv + e * D_, D_, D_, 0, nullptr, vf16);

    k_attn<<<NSEQ * H_, 256, smem_attn, stream>>>(qf16, kf16, vf16, scores,
                                                  ctx16, e > 0 ? 1 : 0);

    gemm(ctx16, wo, bo + e * D_, D_, D_, 1, t32, nullptr);
    k_bnstats<<<D_, 256, 0, stream>>>(hf32, t32, mean, var);
    k_bnapply<<<dim3((unsigned)((MD + 255) / 256)), 256, 0, stream>>>(
        hf32, t32, mean, var, g1 + e * D_, be1 + e * D_, hf32, hf16);

    gemm(hf16, w1, b1 + e * F_, D_, F_, 2, nullptr, ffh16);
    gemm(ffh16, w2, b2 + e * D_, F_, D_, 1, t32, nullptr);
    k_bnstats<<<D_, 256, 0, stream>>>(hf32, t32, mean, var);
    k_bnapply<<<dim3((unsigned)((MD + 255) / 256)), 256, 0, stream>>>(
        hf32, t32, mean, var, g2 + e * D_, be2 + e * D_, hf32, hf16);
  }

  (void)hipMemcpyAsync(d_out, hf32, MD * 4, hipMemcpyDeviceToDevice, stream);
}
